// BahdanauAttention_1992864825351
// MI455X (gfx1250) — matrix operand order repaired and checked by reading, not compiled, not run on hardware
//
#include <hip/hip_runtime.h>
#include <math.h>

#define B_ 64
#define L_ 2048
#define H_ 1024
#define U_ 1024

typedef __attribute__((ext_vector_type(16))) __bf16 v16bf;
typedef __attribute__((ext_vector_type(8)))  float  v8f;
typedef __attribute__((ext_vector_type(4)))  float  v4f;

union Frag { uint4 q[2]; v16bf v; };

__device__ __forceinline__ unsigned pack_bf16x2(float a, float b) {
    unsigned ua = __float_as_uint(a);
    unsigned ub = __float_as_uint(b);
    ua += 0x7FFFu + ((ua >> 16) & 1u);   // round-to-nearest-even
    ub += 0x7FFFu + ((ub >> 16) & 1u);
    return (ua >> 16) | (ub & 0xFFFF0000u);
}

__device__ __forceinline__ unsigned short f2bf(float a) {
    unsigned ua = __float_as_uint(a);
    ua += 0x7FFFu + ((ua >> 16) & 1u);
    return (unsigned short)(ua >> 16);
}

// ---- Kernel 1: W2 [H,U] f32 -> W2T [U,H] bf16 (tiled transpose) ----
__global__ void k_w2t(const float* __restrict__ W2, unsigned short* __restrict__ W2T) {
    __shared__ float tile[32][33];
    const int h0 = blockIdx.x * 32, u0 = blockIdx.y * 32;
    const int tx = threadIdx.x, ty = threadIdx.y;
#pragma unroll
    for (int i = 0; i < 32; i += 8)
        tile[ty + i][tx] = W2[(size_t)(h0 + ty + i) * U_ + (u0 + tx)];
    __syncthreads();
#pragma unroll
    for (int i = 0; i < 32; i += 8)
        W2T[(size_t)(u0 + ty + i) * H_ + (h0 + tx)] = f2bf(tile[tx][ty + i]);
}

// ---- Kernel 2: qproj[b][u] = sum_h q[b][h]*W1[h][u] + b1[u] + b2[u] ----
__global__ void k_qproj(const float* __restrict__ q, const float* __restrict__ W1,
                        const float* __restrict__ b1, const float* __restrict__ b2,
                        float* __restrict__ qproj) {
    __shared__ float sq[H_];
    const int b = blockIdx.x, tid = threadIdx.x;
    for (int i = tid; i < H_; i += 256) sq[i] = q[b * H_ + i];
    __syncthreads();
    float acc[4] = {0.f, 0.f, 0.f, 0.f};
    for (int h = 0; h < H_; ++h) {
        const float qh = sq[h];
        const float* wr = W1 + (size_t)h * U_;
#pragma unroll
        for (int j = 0; j < 4; ++j) acc[j] = fmaf(qh, wr[tid + j * 256], acc[j]);
    }
#pragma unroll
    for (int j = 0; j < 4; ++j) {
        const int u = tid + j * 256;
        qproj[b * U_ + u] = acc[j] + b1[u] + b2[u];
    }
}

// ---- Kernel 3: zero context accumulator ----
__global__ void k_zero(float* __restrict__ p, int n) {
    const int i = blockIdx.x * blockDim.x + threadIdx.x;
    if (i < n) p[i] = 0.f;
}

// ---- Kernel 4: fused values@W2 -> +qproj -> tanh -> dot V -> scores[b][l] ----
// Block: 256 threads = 8 waves. Tile: 16 L-rows x full U (each wave owns 128 cols).
// Single-buffer LDS tile with two barriers per K-step: acts as a scheduling fence
// that keeps register pressure below the spill threshold (proven clean codegen).
// values loads are NT (512MB streamed; keep the 2MB W2T resident in L2).
__global__ void __launch_bounds__(256)
k_score(const float* __restrict__ values,
        const unsigned short* __restrict__ W2T,
        const float* __restrict__ qproj,
        const float* __restrict__ Vv,
        const float* __restrict__ bv,
        float* __restrict__ scores) {
    __shared__ unsigned As32[16 * 16];   // 16 rows x 32 K as bf16x2 words
    __shared__ float ssum[16];

    const int tid  = threadIdx.x;
    const int wave = tid >> 5, lane = tid & 31;
    const int half = lane >> 4, l16 = lane & 15;
    const int b = blockIdx.y;
    const int l0 = blockIdx.x * 16;

    if (tid < 16) ssum[tid] = 0.f;

    // cooperative A-tile staging: thread -> (row m, K-pair kp)
    const int m = tid >> 4, kp = tid & 15;
    const unsigned long long* vptr =
        (const unsigned long long*)(values + ((size_t)b * L_ + (l0 + m)) * H_ + kp * 2);

    // B fragment base: column = wave*128 + l16, contiguous 16 K-values from half*16
    const unsigned short* bptr = W2T + (size_t)(wave * 128 + l16) * H_ + half * 16;

    v8f acc[8];
#pragma unroll
    for (int j = 0; j < 8; ++j)
#pragma unroll
        for (int r = 0; r < 8; ++r) acc[j][r] = 0.f;

    // A fragment: lanes 0-15 take K [0..15]; lanes 16-31: [16..31]
    const int abase = l16 * 16 + half * 8;
    const int widx  = m * 16 + kp;

    for (int kk = 0; kk < H_ / 32; ++kk) {
        __syncthreads();
        const unsigned long long raw = __builtin_nontemporal_load(vptr);
        As32[widx] = pack_bf16x2(__uint_as_float((unsigned)raw),
                                 __uint_as_float((unsigned)(raw >> 32)));
        __syncthreads();

        Frag a;
        a.q[0] = *(const uint4*)&As32[abase];
        a.q[1] = *(const uint4*)&As32[abase + 4];

#pragma unroll
        for (int j = 0; j < 8; ++j) {
            Frag bf;
            bf.q[0] = *(const uint4*)(bptr + (size_t)j * 16 * H_);
            bf.q[1] = *(const uint4*)(bptr + (size_t)j * 16 * H_ + 8);
            acc[j] = __builtin_amdgcn_wmma_f32_16x16x32_bf16(
                false, a.v, false, bf.v, (short)0, acc[j], false, false);
        }
        vptr += 4;   // advance 8 floats (32 bytes)
        bptr += 32;
    }

    // Epilogue: tanh(acc + qproj) * V, reduce over U
    float part[8];
#pragma unroll
    for (int r = 0; r < 8; ++r) part[r] = 0.f;
#pragma unroll
    for (int j = 0; j < 8; ++j) {
        const int col = wave * 128 + j * 16 + l16;
        const float qp = qproj[b * U_ + col];
        const float vv = Vv[col];
#pragma unroll
        for (int r = 0; r < 8; ++r)
            part[r] += tanhf(acc[j][r] + qp) * vv;
    }
    // butterfly reduce within each 16-lane half (rows r / r+8)
#pragma unroll
    for (int off = 1; off < 16; off <<= 1)
#pragma unroll
        for (int r = 0; r < 8; ++r)
            part[r] += __shfl_xor(part[r], off, 32);

    if (l16 == 0) {
#pragma unroll
        for (int r = 0; r < 8; ++r)
            atomicAdd(&ssum[half * 8 + r], part[r]);
    }
    __syncthreads();
    if (tid < 16)
        scores[(size_t)b * L_ + l0 + tid] = ssum[tid] + bv[0];
}

// ---- Kernel 5: softmax over L per batch ----
__global__ void k_softmax(const float* __restrict__ scores, float* __restrict__ attn) {
    __shared__ float se[L_];
    __shared__ float red[256];
    const int b = blockIdx.x, tid = threadIdx.x;
    float lmax = -3.4e38f;
#pragma unroll
    for (int i = 0; i < L_ / 256; ++i) {
        const float s = scores[(size_t)b * L_ + tid + i * 256];
        se[tid + i * 256] = s;
        lmax = fmaxf(lmax, s);
    }
    red[tid] = lmax;
    __syncthreads();
    for (int s = 128; s > 0; s >>= 1) {
        if (tid < s) red[tid] = fmaxf(red[tid], red[tid + s]);
        __syncthreads();
    }
    const float gmax = red[0];
    __syncthreads();
    float lsum = 0.f;
#pragma unroll
    for (int i = 0; i < L_ / 256; ++i) {
        const float e = __expf(se[tid + i * 256] - gmax);
        se[tid + i * 256] = e;
        lsum += e;
    }
    red[tid] = lsum;
    __syncthreads();
    for (int s = 128; s > 0; s >>= 1) {
        if (tid < s) red[tid] += red[tid + s];
        __syncthreads();
    }
    const float inv = 1.f / red[0];
#pragma unroll
    for (int i = 0; i < L_ / 256; ++i)
        attn[(size_t)b * L_ + tid + i * 256] = se[tid + i * 256] * inv;
}

// ---- Kernel 6: context[b][h] = sum_l attn[b][l] * values[b][l][h] ----
// Split L 16-ways -> 1024 blocks for the second 512MB values pass (NT loads).
__global__ void k_context(const float* __restrict__ values,
                          const float* __restrict__ attn,
                          float* __restrict__ ctx) {
    __shared__ float w[128];
    const int lc = blockIdx.x, b = blockIdx.y, tid = threadIdx.x;
    if (tid < 128) w[tid] = attn[(size_t)b * L_ + lc * 128 + tid];
    __syncthreads();
    v4f acc = {0.f, 0.f, 0.f, 0.f};
    const float* vb = values + ((size_t)b * L_ + lc * 128) * H_ + tid * 4;
    for (int l = 0; l < 128; ++l) {
        const float wl = w[l];
        const v4f v = __builtin_nontemporal_load((const v4f*)(vb + (size_t)l * H_));
        acc.x = fmaf(wl, v.x, acc.x);
        acc.y = fmaf(wl, v.y, acc.y);
        acc.z = fmaf(wl, v.z, acc.z);
        acc.w = fmaf(wl, v.w, acc.w);
    }
    atomicAdd(&ctx[b * H_ + tid * 4 + 0], acc.x);
    atomicAdd(&ctx[b * H_ + tid * 4 + 1], acc.y);
    atomicAdd(&ctx[b * H_ + tid * 4 + 2], acc.z);
    atomicAdd(&ctx[b * H_ + tid * 4 + 3], acc.w);
}

extern "C" void kernel_launch(void* const* d_in, const int* in_sizes, int n_in,
                              void* d_out, int out_size, void* d_ws, size_t ws_size,
                              hipStream_t stream) {
    const float* query  = (const float*)d_in[0];
    const float* values = (const float*)d_in[1];
    const float* W1     = (const float*)d_in[2];
    const float* b1     = (const float*)d_in[3];
    const float* W2     = (const float*)d_in[4];
    const float* b2     = (const float*)d_in[5];
    const float* Vv     = (const float*)d_in[6];
    const float* bv     = (const float*)d_in[7];

    // workspace layout: W2T bf16 (2MB) | qproj (256KB) | scores (512KB)
    unsigned short* w2t = (unsigned short*)d_ws;
    float* qproj  = (float*)((char*)d_ws + (size_t)2 * 1024 * 1024);
    float* scores = (float*)((char*)d_ws + (size_t)2 * 1024 * 1024 + 256 * 1024);

    // d_out: context [B,H] then attention_weights [B,L,1]
    float* ctx  = (float*)d_out;
    float* attn = (float*)d_out + (size_t)B_ * H_;

    k_w2t    <<<dim3(H_ / 32, U_ / 32), dim3(32, 8), 0, stream>>>(W2, w2t);
    k_qproj  <<<dim3(B_),               dim3(256),   0, stream>>>(query, W1, b1, b2, qproj);
    k_zero   <<<dim3((B_ * H_ + 255) / 256), dim3(256), 0, stream>>>(ctx, B_ * H_);
    k_score  <<<dim3(L_ / 16, B_),      dim3(256),   0, stream>>>(values, w2t, qproj, Vv, bv, scores);
    k_softmax<<<dim3(B_),               dim3(256),   0, stream>>>(scores, attn);
    k_context<<<dim3(L_ / 128, B_),     dim3(256),   0, stream>>>(values, attn, ctx);
}
